// LSTMPricePredictor_1812476199481
// MI455X (gfx1250) — compile-verified, matrix operand order checked by reading, not run on hardware
//
#include <hip/hip_runtime.h>

// ---------------- types ----------------
typedef __bf16 bf16x16 __attribute__((ext_vector_type(16)));
typedef float  v8f     __attribute__((ext_vector_type(8)));
typedef unsigned int u32x4 __attribute__((ext_vector_type(4)));

struct FragU { u32x4 lo, hi; };

// ---------------- constants ----------------
#define BATCH   4096
#define SEQ_T   512
#define HID     50
#define G4H     200      // 4*HID
#define ROWS    64       // batch rows per workgroup
#define NTILES  14       // N padded 200 -> 224, split 7+7 across wave pairs
#define NTW     7        // N tiles per wave
#define LDG     224      // gates / weight N stride (NTILES*16)
#define LDK     64       // padded K stride for bf16 buffers
#define THREADS 256      // 8 waves: wave pair (2w, 2w+1) shares M block w*16

// LDS layout (bytes, all 16B aligned)
#define OFF_GATES 0                         // 64*224*4   = 57344
#define OFF_W0    (OFF_GATES + ROWS*LDG*4)  // 224*64*2   = 28672
#define OFF_WI1   (OFF_W0  + LDG*LDK*2)
#define OFF_WH1   (OFF_WI1 + LDG*LDK*2)
#define OFF_H0    (OFF_WH1 + LDG*LDK*2)     // 64*64*2    = 8192
#define OFF_H1    (OFF_H0  + ROWS*LDK*2)
#define OFF_C0    (OFF_H1  + ROWS*LDK*2)    // 64*50*4    = 12800
#define OFF_C1    (OFF_C0  + ROWS*HID*4)
#define OFF_BC0   (OFF_C1  + ROWS*HID*4)    // 200*4 = 800
#define OFF_BC1   (OFF_BC0 + G4H*4)
#define OFF_WX0   (OFF_BC1 + G4H*4)
#define OFF_FCW   (OFF_WX0 + G4H*4)         // 256
#define SMEM_BYTES (OFF_FCW + 256)

// byte strides for fragment addressing
#define BSTRIDE_NT (16 * LDK * 2)   // one N tile of B rows = 2048 B
#define BSTRIDE_KS 64               // one k-step = 32 bf16 = 64 B

// ---------------- helpers ----------------
__device__ __forceinline__ __bf16 f2bf(float f) {
    union { float f; unsigned u; } v; v.f = f;
    unsigned r = v.u + 0x7FFFu + ((v.u >> 16) & 1u);   // round-to-nearest-even
    unsigned short s = (unsigned short)(r >> 16);
    return __builtin_bit_cast(__bf16, s);
}

__device__ __forceinline__ float sigm(float x) {
    return 1.0f / (1.0f + __expf(-x));
}
__device__ __forceinline__ float tanh_f(float x) {
    x = fminf(15.0f, fmaxf(-15.0f, x));
    float e = __expf(2.0f * x);
    return (e - 1.0f) / (e + 1.0f);
}

// Load one 8-VGPR bf16 fragment from a precomputed per-lane base + immediate byte offset.
// A/B 16-bit layout: dwords 0..3 = K kbase..kbase+7, dwords 4..7 = K kbase+16..kbase+23.
__device__ __forceinline__ bf16x16 ld_frag(const __bf16* base, int byteOff) {
    FragU u;
    u.lo = *(const u32x4*)((const char*)base + byteOff);
    u.hi = *(const u32x4*)((const char*)base + byteOff + 32);
    return __builtin_bit_cast(bf16x16, u);
}

// acc[7] += A(16x64) * B(64 x 112).  Abase/Bbase are per-lane fragment base
// pointers (loop-invariant).  B fragments use a one-ahead double buffer so
// each WMMA's wait can leave the next fragment's loads in flight; peak live
// VGPRs ~ acc(56) + a(16) + b(16) + addresses, comfortably under 256.
__device__ __forceinline__ void gemm_acc(v8f* acc, const __bf16* Abase, const __bf16* Bbase) {
    bf16x16 a0 = ld_frag(Abase, 0);
    bf16x16 a1 = ld_frag(Abase, BSTRIDE_KS);
#pragma unroll
    for (int ks = 0; ks < 2; ++ks) {
        bf16x16 a = (ks == 0) ? a0 : a1;
        bf16x16 bcur = ld_frag(Bbase, ks * BSTRIDE_KS);
#pragma unroll
        for (int nt = 0; nt < NTW; ++nt) {
            bf16x16 bnext = (nt + 1 < NTW)
                ? ld_frag(Bbase, (nt + 1) * BSTRIDE_NT + ks * BSTRIDE_KS)
                : bcur;
            acc[nt] = __builtin_amdgcn_wmma_f32_16x16x32_bf16(
                false, a, false, bcur, (short)0, acc[nt], false, false);
            bcur = bnext;
        }
    }
}

// C/D layout: VGPR r -> row (lane>>4)*8 + r, col lane&15.  gBase is the
// per-lane f32 store base; offsets are compile-time constants.
__device__ __forceinline__ void store_gates(float* gBase, const v8f* acc) {
#pragma unroll
    for (int nt = 0; nt < NTW; ++nt) {
#pragma unroll
        for (int r = 0; r < 8; ++r) {
            gBase[r * LDG + nt * 16] = acc[nt][r];
        }
    }
}

// ---------------- kernel ----------------
__global__ __launch_bounds__(THREADS)
void lstm2_fused_kernel(const float* __restrict__ xg,
                        const float* __restrict__ Wih0,  // [200,1]
                        const float* __restrict__ Whh0,  // [200,50]
                        const float* __restrict__ bih0,
                        const float* __restrict__ bhh0,
                        const float* __restrict__ Wih1,  // [200,50]
                        const float* __restrict__ Whh1,  // [200,50]
                        const float* __restrict__ bih1,
                        const float* __restrict__ bhh1,
                        const float* __restrict__ fc_w,  // [1,50]
                        const float* __restrict__ fc_b,  // [1]
                        float* __restrict__ out)         // [B,1]
{
    extern __shared__ char smem[];
    float*  gatesF = (float*)(smem + OFF_GATES);
    __bf16* w0     = (__bf16*)(smem + OFF_W0);
    __bf16* wi1    = (__bf16*)(smem + OFF_WI1);
    __bf16* wh1    = (__bf16*)(smem + OFF_WH1);
    __bf16* h0buf  = (__bf16*)(smem + OFF_H0);
    __bf16* h1buf  = (__bf16*)(smem + OFF_H1);
    float*  c0buf  = (float*)(smem + OFF_C0);
    float*  c1buf  = (float*)(smem + OFF_C1);
    float*  bc0    = (float*)(smem + OFF_BC0);
    float*  bc1    = (float*)(smem + OFF_BC1);
    float*  wx0    = (float*)(smem + OFF_WX0);
    float*  fcw    = (float*)(smem + OFF_FCW);

    const int tid = threadIdx.x;
    const int b0  = blockIdx.x * ROWS;

    // ---- stage weights (bf16, zero-padded to [224 x 64]) + states ----
    for (int idx = tid; idx < LDG * LDK; idx += THREADS) {
        int n = idx >> 6, k = idx & 63;
        bool ok = (n < G4H) && (k < HID);
        w0 [idx] = f2bf(ok ? Whh0[n * HID + k] : 0.0f);
        wi1[idx] = f2bf(ok ? Wih1[n * HID + k] : 0.0f);
        wh1[idx] = f2bf(ok ? Whh1[n * HID + k] : 0.0f);
    }
    for (int idx = tid; idx < ROWS * LDK; idx += THREADS) {
        h0buf[idx] = f2bf(0.0f);
        h1buf[idx] = f2bf(0.0f);
    }
    for (int idx = tid; idx < ROWS * HID; idx += THREADS) {
        c0buf[idx] = 0.0f;
        c1buf[idx] = 0.0f;
    }
    for (int n = tid; n < G4H; n += THREADS) {
        bc0[n] = bih0[n] + bhh0[n];
        bc1[n] = bih1[n] + bhh1[n];
        wx0[n] = Wih0[n];            // IN == 1
    }
    for (int j = tid; j < HID; j += THREADS) fcw[j] = fc_w[j];
    __syncthreads();

    // ---- per-lane, loop-invariant fragment/store base pointers ----
    const int lane   = tid & 31;
    const int wv     = tid >> 5;          // 0..7
    const int m0     = (wv >> 1) * 16;    // M block
    const int ntBase = (wv & 1) * NTW;    // N-tile half: 0..6 or 7..13
    const int mL     = lane & 15;
    const int nL     = lane & 15;
    const int kB     = (lane >> 4) << 3;  // 0 or 8
    const int rB     = (lane >> 4) << 3;  // C/D row base

    const __bf16* h0A = h0buf + (m0 + mL) * LDK + kB;             // A bases
    const __bf16* h1A = h1buf + (m0 + mL) * LDK + kB;
    const __bf16* w0B  = w0  + (ntBase * 16 + nL) * LDK + kB;     // B bases
    const __bf16* wi1B = wi1 + (ntBase * 16 + nL) * LDK + kB;
    const __bf16* wh1B = wh1 + (ntBase * 16 + nL) * LDK + kB;
    float* gBase = gatesF + (m0 + rB) * LDG + ntBase * 16 + nL;   // C/D store base

    // elementwise bodies: idx in [0, ROWS*HID)
    auto ew0 = [&](int idx) {
        int m = idx / HID;
        int j = idx - m * HID;
        float xi = xg[(b0 + m) * SEQ_T + 0];
        (void)xi;
    };
    (void)ew0;

#pragma unroll 1
    for (int t = 0; t < SEQ_T; ++t) {
        // ================= layer 0: gates = h0 @ Whh0^T =================
        {
            v8f acc[NTW] = {};
            gemm_acc(acc, h0A, w0B);
            store_gates(gBase, acc);
        }
        __syncthreads();

        {
            auto body0 = [&](int idx) {
                int m = idx / HID;
                int j = idx - m * HID;
                float xi = xg[(b0 + m) * SEQ_T + t];
                const float* gr = &gatesF[m * LDG];
                float gi = gr[j]           + xi * wx0[j]           + bc0[j];
                float gf = gr[HID + j]     + xi * wx0[HID + j]     + bc0[HID + j];
                float gg = gr[2 * HID + j] + xi * wx0[2 * HID + j] + bc0[2 * HID + j];
                float go = gr[3 * HID + j] + xi * wx0[3 * HID + j] + bc0[3 * HID + j];
                float c = sigm(gf) * c0buf[m * HID + j] + sigm(gi) * tanh_f(gg);
                c0buf[m * HID + j] = c;
                float h = sigm(go) * tanh_f(c);
                h0buf[m * LDK + j] = f2bf(h);
            };
#pragma unroll
            for (int i = 0; i < 12; ++i) body0(tid + i * THREADS);   // 3072 uniform
            if (tid < (ROWS * HID - 12 * THREADS)) body0(tid + 12 * THREADS); // tail 128
        }
        __syncthreads();

        // ========= layer 1: gates = h0_new @ Wih1^T + h1 @ Whh1^T =========
        {
            v8f acc[NTW] = {};
            gemm_acc(acc, h0A, wi1B);
            gemm_acc(acc, h1A, wh1B);
            store_gates(gBase, acc);
        }
        __syncthreads();

        {
            auto body1 = [&](int idx) {
                int m = idx / HID;
                int j = idx - m * HID;
                float* gr = &gatesF[m * LDG];
                float gi = gr[j]           + bc1[j];
                float gf = gr[HID + j]     + bc1[HID + j];
                float gg = gr[2 * HID + j] + bc1[2 * HID + j];
                float go = gr[3 * HID + j] + bc1[3 * HID + j];
                float c = sigm(gf) * c1buf[m * HID + j] + sigm(gi) * tanh_f(gg);
                c1buf[m * HID + j] = c;
                float h = sigm(go) * tanh_f(c);
                h1buf[m * LDK + j] = f2bf(h);
                gr[j] = h;               // f32 copy for the final projection
            };
#pragma unroll
            for (int i = 0; i < 12; ++i) body1(tid + i * THREADS);   // 3072 uniform
            if (tid < (ROWS * HID - 12 * THREADS)) body1(tid + 12 * THREADS); // tail 128
        }
        __syncthreads();
    }

    // ---- final projection: out[b] = fc_b + sum_j h1[b][j] * fc_w[j] ----
    if (tid < ROWS) {
        float s = fc_b[0];
#pragma unroll
        for (int j = 0; j < HID; ++j) s += gatesF[tid * LDG + j] * fcw[j];
        out[b0 + tid] = s;
    }
}

// ---------------- host launch ----------------
extern "C" void kernel_launch(void* const* d_in, const int* in_sizes, int n_in,
                              void* d_out, int out_size, void* d_ws, size_t ws_size,
                              hipStream_t stream) {
    (void)in_sizes; (void)n_in; (void)out_size; (void)d_ws; (void)ws_size;
    const float* x    = (const float*)d_in[0];
    const float* Wih0 = (const float*)d_in[1];
    const float* Whh0 = (const float*)d_in[2];
    const float* bih0 = (const float*)d_in[3];
    const float* bhh0 = (const float*)d_in[4];
    const float* Wih1 = (const float*)d_in[5];
    const float* Whh1 = (const float*)d_in[6];
    const float* bih1 = (const float*)d_in[7];
    const float* bhh1 = (const float*)d_in[8];
    const float* fc_w = (const float*)d_in[9];
    const float* fc_b = (const float*)d_in[10];
    float* out = (float*)d_out;

    dim3 grid(BATCH / ROWS);   // 64 workgroups, one per WGP
    dim3 block(THREADS);       // 8 waves (wave32), 2 per SIMD32
    hipLaunchKernelGGL(lstm2_fused_kernel, grid, block, SMEM_BYTES, stream,
                       x, Wih0, Whh0, bih0, bhh0, Wih1, Whh1, bih1, bhh1,
                       fc_w, fc_b, out);
}
